// KalmanFormerNet_14559939133440
// MI455X (gfx1250) — matrix operand order repaired and checked by reading, not compiled, not run on hardware
//
#include <hip/hip_runtime.h>
#include <hip/hip_bf16.h>
#include <stdint.h>

#define B_N 8
#define T_N 16
#define L_N 1024
#define D_N 256
#define K_N 128
#define KD_N (K_N * D_N)          /* 32768 */
#define STATE_N (B_N * KD_N)      /* 262144 */

typedef __attribute__((ext_vector_type(16))) __bf16 bf16x16;
typedef __attribute__((ext_vector_type(8)))  float  f32x8;

union FragU { uint4 q[2]; bf16x16 v; };

__device__ __forceinline__ uint16_t f2bf(float f) {
  uint32_t u = __float_as_uint(f);
  u += 0x7FFFu + ((u >> 16) & 1u);        // round-to-nearest-even
  return (uint16_t)(u >> 16);
}
__device__ __forceinline__ uint32_t pack2bf(float lo, float hi) {
  return (uint32_t)f2bf(lo) | ((uint32_t)f2bf(hi) << 16);
}
__device__ __forceinline__ float gelu_tanh(float x) {
  float x3 = x * x * x;
  return 0.5f * x * (1.0f + tanhf(0.7978845608f * (x + 0.044715f * x3)));
}

// ---------------------------------------------------------------------------
// prep: zero recurrence state + pack all weights (f32 row-major -> bf16 WMMA
// B-fragment tile layout: tile(kt,nt) = 32 lanes x 16 contiguous bf16, lane
// holds K = 32*kt + 16*(lane>>4) + [0..15] at column 16*nt + (lane&15)).
// ---------------------------------------------------------------------------
__device__ __forceinline__ void pack_one(const float* __restrict__ W,
                                         uint32_t* __restrict__ dst,
                                         int Kd, int tid, int nth) {
  const int total = (Kd * D_N) >> 1;      // dwords
  for (int d = tid; d < total; d += nth) {
    int v = d & 7, lane = (d >> 3) & 31, tile = d >> 8;
    int nt = tile & 15, kt = tile >> 4, h = lane >> 4;
    int k   = (kt << 5) + (h << 4) + (v << 1);
    int col = (nt << 4) + (lane & 15);
    dst[d] = pack2bf(W[k * D_N + col], W[(k + 1) * D_N + col]);
  }
}

__global__ void prep_kernel(const float* __restrict__ We, const float* __restrict__ Wq,
                            const float* __restrict__ Wk, const float* __restrict__ Wv,
                            const float* __restrict__ Wo,
                            uint32_t* __restrict__ wpack, float* __restrict__ state3) {
  int tid = blockIdx.x * blockDim.x + threadIdx.x;
  int nth = gridDim.x * blockDim.x;
  for (int i = tid; i < 3 * STATE_N; i += nth) state3[i] = 0.0f;  // vel/prev_innov/prev_sud
  pack_one(We, wpack + 0,      512, tid, nth);
  pack_one(Wq, wpack + 65536,  256, tid, nth);
  pack_one(Wk, wpack + 98304,  512, tid, nth);
  pack_one(Wv, wpack + 163840, 512, tid, nth);
  pack_one(Wo, wpack + 229376, 256, tid, nth);
}

// ---------------------------------------------------------------------------
// signal + top-k + gather:   one workgroup per (b,t)
// ---------------------------------------------------------------------------
__global__ void __launch_bounds__(256) signal_topk_kernel(
    const float* __restrict__ x, const float* __restrict__ wq,
    const float* __restrict__ wk, float* __restrict__ out,
    float* __restrict__ assoc0) {
  __shared__ float sc[L_N];
  __shared__ int   si[L_N];
  __shared__ float swq[D_N], swk[D_N];
  const int tid = threadIdx.x;
  const int b = blockIdx.x / T_N, t = blockIdx.x % T_N;
  const int tp = (t + T_N - 1) % T_N;           // jnp.roll(x, 1, axis=1)
  swq[tid] = wq[tid];
  swk[tid] = wk[tid];
  __syncthreads();
  for (int l = tid; l < L_N; l += 256) {
    const float* xq = x + ((size_t)((b * T_N + t)  * L_N + l)) * D_N;
    const float* xk = x + ((size_t)((b * T_N + tp) * L_N + l)) * D_N;
    float q = 0.f, k = 0.f;
    for (int d = 0; d < D_N; ++d) { q += xq[d] * swq[d]; k += xk[d] * swk[d]; }
    sc[l] = q * k * 0.0625f;                    // 1/sqrt(256)
    si[l] = l;
  }
  // bitonic sort, descending by value, ascending index tie-break
  for (int ksz = 2; ksz <= L_N; ksz <<= 1) {
    for (int j = ksz >> 1; j > 0; j >>= 1) {
      __syncthreads();
      for (int t2 = tid; t2 < (L_N >> 1); t2 += 256) {
        int i = ((t2 / j) * (j << 1)) + (t2 % j);
        int p = i + j;
        bool desc = ((i & ksz) == 0);
        float a = sc[i], c = sc[p];
        int   ia = si[i], ic = si[p];
        bool gt = (a > c) || (a == c && ia < ic);
        if (gt != desc) { sc[i] = c; sc[p] = a; si[i] = ic; si[p] = ia; }
      }
    }
  }
  __syncthreads();
  // gather selected tokens straight into out[b,t]; also seed assoc0 = sel[:,0]
  float*       ob = out + ((size_t)((b * T_N + t) * K_N)) * D_N;
  const float* xb = x   + ((size_t)((b * T_N + t) * L_N)) * D_N;
  for (int kk = 0; kk < K_N; ++kk) {
    int idx = si[kk];
    float val = xb[(size_t)idx * D_N + tid];
    ob[(size_t)kk * D_N + tid] = val;
    if (t == 0) assoc0[b * KD_N + kk * D_N + tid] = val;
  }
}

// ---------------------------------------------------------------------------
// WMMA GEMM helper: 8 waves, wave w owns output rows [16w,16w+16), loops
// over `ntiles` 16-col tiles, K in chunks of 32 (v_wmma_f32_16x16x32_bf16).
// A: row-major bf16 in LDS (2x ds_load_b128 per fragment).
// B: WGLOBAL -> pre-packed global weight tiles (2x global_load_b128/lane);
//    else    -> LDS matrix laid out so B[k][n] = Bsrc[n*ldb + k] (contiguous k).
// ---------------------------------------------------------------------------
template <bool WGLOBAL, class EpiFn>
__device__ __forceinline__ void run_gemm(const uint16_t* __restrict__ A, int lda, int Kd,
                                         const uint16_t* __restrict__ Bsrc, int ldb,
                                         int ntiles, EpiFn epi) {
  const int lane = threadIdx.x & 31;
  const int wave = threadIdx.x >> 5;
  const int h    = lane >> 4;
  const int arow = (wave << 4) + (lane & 15);
  const int kt_n = Kd >> 5;
  const uint16_t* Ar = A + arow * lda;
  for (int nt = 0; nt < ntiles; ++nt) {
    f32x8 acc = {0.f, 0.f, 0.f, 0.f, 0.f, 0.f, 0.f, 0.f};
    for (int kt = 0; kt < kt_n; ++kt) {
      const int k0 = kt << 5;
      FragU a, bfr;
      a.q[0] = *(const uint4*)(Ar + k0 + 16 * h);         // K = k0+16h .. +7
      a.q[1] = *(const uint4*)(Ar + k0 + 16 * h + 8);     // K = k0+16h+8 .. +15
      const uint16_t* bp;
      if constexpr (WGLOBAL) {
        bp = Bsrc + (size_t)(((kt * 16 + nt) << 5) + lane) * 16;
      } else {
        bp = Bsrc + ((nt << 4) + (lane & 15)) * ldb + k0 + (h << 4);
      }
      bfr.q[0] = *(const uint4*)(bp);                      // K = k0+16h .. +7
      bfr.q[1] = *(const uint4*)(bp + 8);                  // K = k0+16h+8 .. +15
      acc = __builtin_amdgcn_wmma_f32_16x16x32_bf16(false, a.v, false, bfr.v,
                                                    (short)0, acc, false, false);
    }
    epi(acc, nt);
  }
}

// ---------------------------------------------------------------------------
// persistent Kalman recurrence: one workgroup per batch, loops t = 1..15.
// LDS (320 KB): bufA 128x512 bf16 | kh 128x256 | vh^T 256x128 | h/scores 64KB
// ---------------------------------------------------------------------------
__global__ void __launch_bounds__(256) recur_kernel(
    float* __restrict__ out, const uint16_t* __restrict__ wpack,
    float* __restrict__ stf, const float* __restrict__ b_enc,
    const float* __restrict__ b_out) {
  extern __shared__ char smem_raw[];
  uint16_t* smem  = (uint16_t*)smem_raw;
  uint16_t* bufA  = smem;              // 65536 elems: dec_kv/enc_in; [0,32768)=qh/ctx
  uint16_t* bufKH = smem + 65536;      // 32768 elems: kh row-major [m][d]
  uint16_t* bufVH = smem + 98304;      // 32768 elems: vh transposed [d][m]
  uint16_t* bufH  = smem + 131072;     // 32768 elems: h bf16 / scores f32 (64 KB)
  float*    bufSf = (float*)bufH;
  uint16_t* bufP  = bufA + 32768;      // 128x128 bf16 softmax probs

  const int tid  = threadIdx.x;
  const int lane = tid & 31, wave = tid >> 5;
  const int h = lane >> 4, c15 = lane & 15;
  const int wbase = wave << 4;
  const int b  = blockIdx.x;
  const int bo = b * KD_N;

  float* vel    = stf;
  float* pinov  = stf + STATE_N;
  float* psud   = stf + 2 * STATE_N;
  float* predA  = stf + 3 * STATE_N;
  float* cinov  = stf + 4 * STATE_N;
  float* assoc0 = stf + 5 * STATE_N;
  float* assoc1 = stf + 6 * STATE_N;

  const uint16_t* w_enc = wpack;
  const uint16_t* w_q   = wpack + 131072;
  const uint16_t* w_k   = wpack + 196608;
  const uint16_t* w_v   = wpack + 327680;
  const uint16_t* w_o   = wpack + 458752;

  // prefetch the 1 MB packed-weight block (stays L2/L1-resident for all steps)
  for (int ln = tid; ln < 16384; ln += 256)
    __builtin_prefetch((const char*)wpack + (size_t)ln * 64, 0, 3);

  for (int t = 1; t < T_N; ++t) {
    float* assocP = ((t - 1) & 1) ? assoc1 : assoc0;   // holds sel[:,t-1]
    float* assocN = ((t - 1) & 1) ? assoc0 : assoc1;
    float* obs = out + ((size_t)(b * T_N + t)) * KD_N; // sel[:,t] (not yet overwritten)

    // pass 1: elementwise state + dec_kv = [evol_diff | prev_sud] -> bufA (bf16)
    for (int i = tid; i < KD_N; i += 256) {
      int row = i >> 8, col = i & 255;
      float o  = obs[i];
      float pa = assocP[bo + i];
      float v  = vel[bo + i];
      float pi = pinov[bo + i];
      float ps = psud[bo + i];
      float pr = pa + v;                 // pred
      float in = o - pr;                 // innov
      predA[bo + i]  = pr;
      cinov[bo + i]  = in;
      pinov[bo + i]  = in;               // becomes prev_innov for t+1
      assocN[bo + i] = o;                // becomes prev_assoc for t+1
      bufA[row * 512 + col]       = f2bf(in - pi);   // evol_diff
      bufA[row * 512 + 256 + col] = f2bf(ps);        // prev_sud
    }
    __syncthreads();

    // kh = dec_kv @ W_k ; vh = dec_kv @ W_v (stored transposed for ctx GEMM)
    run_gemm<true>(bufA, 512, 512, w_k, 0, 16, [&](const f32x8& a, int nt) {
#pragma unroll
      for (int r = 0; r < 8; ++r) {
        int row = wbase + r + 8 * h, col = (nt << 4) + c15;
        bufKH[row * 256 + col] = f2bf(a[r]);
      }
    });
    run_gemm<true>(bufA, 512, 512, w_v, 0, 16, [&](const f32x8& a, int nt) {
#pragma unroll
      for (int r = 0; r < 8; ++r) {
        int row = wbase + r + 8 * h, col = (nt << 4) + c15;
        bufVH[col * 128 + row] = f2bf(a[r]);
      }
    });
    __syncthreads();

    // pass 2: enc_in = [innov | vel] -> bufA
    for (int i = tid; i < KD_N; i += 256) {
      int row = i >> 8, col = i & 255;
      bufA[row * 512 + col]       = f2bf(cinov[bo + i]);
      bufA[row * 512 + 256 + col] = f2bf(vel[bo + i]);
    }
    __syncthreads();

    // h = gelu(enc_in @ W_enc + b_enc)
    run_gemm<true>(bufA, 512, 512, w_enc, 0, 16, [&](const f32x8& a, int nt) {
#pragma unroll
      for (int r = 0; r < 8; ++r) {
        int row = wbase + r + 8 * h, col = (nt << 4) + c15;
        bufH[row * 256 + col] = f2bf(gelu_tanh(a[r] + b_enc[col]));
      }
    });
    __syncthreads();

    // qh = h @ W_q  -> bufA[0,32768)
    run_gemm<true>(bufH, 256, 256, w_q, 0, 16, [&](const f32x8& a, int nt) {
#pragma unroll
      for (int r = 0; r < 8; ++r) {
        int row = wbase + r + 8 * h, col = (nt << 4) + c15;
        bufA[row * 256 + col] = f2bf(a[r]);
      }
    });
    __syncthreads();

    // scores = qh @ kh^T * inv_sqrt_d  (f32 into bufSf, overwrites h)
    run_gemm<false>(bufA, 256, 256, bufKH, 256, 8, [&](const f32x8& a, int nt) {
#pragma unroll
      for (int r = 0; r < 8; ++r) {
        int row = wbase + r + 8 * h, col = (nt << 4) + c15;
        bufSf[row * 128 + col] = a[r] * 0.0625f;
      }
    });
    __syncthreads();

    // softmax per row -> P (bf16)
    if (tid < 128) {
      const float* Sr = bufSf + tid * 128;
      float m = -3.402823e38f;
      for (int c = 0; c < 128; ++c) m = fmaxf(m, Sr[c]);
      float s = 0.f;
      for (int c = 0; c < 128; ++c) s += __expf(Sr[c] - m);
      float inv = 1.0f / s;
      for (int c = 0; c < 128; ++c) bufP[tid * 128 + c] = f2bf(__expf(Sr[c] - m) * inv);
    }
    __syncthreads();

    // ctx = P @ vh  -> bufA[0,32768)  (A = bufP, disjoint from write region)
    run_gemm<false>(bufP, 128, 128, bufVH, 128, 16, [&](const f32x8& a, int nt) {
#pragma unroll
      for (int r = 0; r < 8; ++r) {
        int row = wbase + r + 8 * h, col = (nt << 4) + c15;
        bufA[row * 256 + col] = f2bf(a[r]);
      }
    });
    __syncthreads();

    // Kg = sigmoid(ctx @ W_out + b_out); fused Kalman update epilogue
    run_gemm<true>(bufA, 256, 256, w_o, 0, 16, [&](const f32x8& a, int nt) {
#pragma unroll
      for (int r = 0; r < 8; ++r) {
        int row = wbase + r + 8 * h, col = (nt << 4) + c15;
        int i = row * 256 + col;
        float g  = 1.0f / (1.0f + __expf(-(a[r] + b_out[col])));
        float in = cinov[bo + i];
        float pr = predA[bo + i];
        float xp = pr + g * in;
        obs[i]       = xp;                    // posts -> out[b,t]
        psud[bo + i] = g * in;                // sud = x_post - pred
        vel[bo + i]  = xp - assocP[bo + i];   // new velocity
      }
    });
    __syncthreads();
  }
}

// ---------------------------------------------------------------------------
extern "C" void kernel_launch(void* const* d_in, const int* in_sizes, int n_in,
                              void* d_out, int out_size, void* d_ws, size_t ws_size,
                              hipStream_t stream) {
  (void)in_sizes; (void)n_in; (void)out_size; (void)ws_size;
  const float* x      = (const float*)d_in[0];
  const float* w_qsig = (const float*)d_in[1];
  const float* w_ksig = (const float*)d_in[2];
  const float* W_enc  = (const float*)d_in[3];
  const float* b_enc  = (const float*)d_in[4];
  const float* W_q    = (const float*)d_in[5];
  const float* W_k    = (const float*)d_in[6];
  const float* W_v    = (const float*)d_in[7];
  const float* W_out  = (const float*)d_in[8];
  const float* b_out  = (const float*)d_in[9];
  float* out = (float*)d_out;

  uint32_t* wpack = (uint32_t*)d_ws;                       // 262144 dwords packed bf16
  float* stf = (float*)((char*)d_ws + (size_t)262144 * 4); // 7 state arrays x 1 MB
  float* assoc0 = stf + 5 * STATE_N;

  prep_kernel<<<512, 256, 0, stream>>>(W_enc, W_q, W_k, W_v, W_out, wpack, stf);
  signal_topk_kernel<<<B_N * T_N, 256, 0, stream>>>(x, w_qsig, w_ksig, out, assoc0);

  hipFuncSetAttribute((const void*)recur_kernel,
                      hipFuncAttributeMaxDynamicSharedMemorySize, 327680);
  recur_kernel<<<B_N, 256, 327680, stream>>>(out, (const uint16_t*)d_ws, stf,
                                             b_enc, b_out);
}
